// GraphAttentionNetworkInductive_20392504721832
// MI455X (gfx1250) — matrix operand order repaired and checked by reading, not compiled, not run on hardware
//
#include <hip/hip_runtime.h>
#include <stdint.h>

typedef __attribute__((ext_vector_type(16))) _Float16 v16h;
typedef __attribute__((ext_vector_type(8)))  float    v8f;

#define NEG_SLOPE 0.2f

static inline int ceil_div_i(int a, int b) { return (a + b - 1) / b; }

// ---------------- utility kernels ----------------
__global__ void fill_f32_kernel(float* p, float v, int n) {
  int i = blockIdx.x * blockDim.x + threadIdx.x;
  if (i < n) p[i] = v;
}

__global__ void fill_i32_kernel(int* p, int v, int n) {
  int i = blockIdx.x * blockDim.x + threadIdx.x;
  if (i < n) p[i] = v;
}

__global__ void cvt_f32_to_f16_kernel(const float* __restrict__ s, _Float16* __restrict__ d, int n) {
  int i = blockIdx.x * blockDim.x + threadIdx.x;
  if (i < n) d[i] = (_Float16)s[i];
}

// Convert weight [K,N] f32 -> [K,Npad] f16 (zero padded columns).
__global__ void cvt_pad_f16_kernel(const float* __restrict__ s, _Float16* __restrict__ d,
                                   int K, int N, int Npad) {
  int i = blockIdx.x * blockDim.x + threadIdx.x;
  if (i >= K * Npad) return;
  int k = i / Npad, n = i % Npad;
  d[i] = (n < N) ? (_Float16)s[(size_t)k * N + n] : (_Float16)0.0f;
}

// ---------------- WMMA GEMM: C[M,Nout] = A[M,K](f16) * B[K,ldb](f16), fp32 out ----------------
// 128x128 tile per 256-thread block; 8 waves in 4(M) x 2(N); wave tile 32x64 = 2x4 fragments.
// A tile staged with GLOBAL_LOAD_ASYNC_TO_LDS_B128 (ASYNCcnt); B staged + transposed via VGPRs.
__global__ __launch_bounds__(256)
void gemm_f16_wmma_kernel(const _Float16* __restrict__ A, const _Float16* __restrict__ B,
                          float* __restrict__ C, int M, int Nout, int K, int ldb, int ldc) {
  __shared__ _Float16 As[128][40];   // [m][k], row start 80B-aligned
  __shared__ _Float16 Bs[128][40];   // [n][k] (transposed)

  const int tid   = threadIdx.x;
  const int lane  = tid & 31;
  const int wave  = tid >> 5;
  const int waveM = wave >> 1;   // 0..3
  const int waveN = wave & 1;    // 0..1
  const int bm    = blockIdx.x * 128;
  const int bn    = blockIdx.y * 128;
  const int lm    = lane & 15;   // position within 16
  const int hs    = lane >> 4;   // half-wave select

  v8f zero = {};
  v8f acc[2][4];
  for (int i = 0; i < 2; ++i)
    for (int j = 0; j < 4; ++j) acc[i][j] = zero;

  union V16 { v16h v; uint4 q[2]; };

  for (int k0 = 0; k0 < K; k0 += 32) {
    // Stage A tile: 128 rows x 32 halves = 512 chunks of 16B, async copy global->LDS.
    // OOB rows are clamped to M-1: they only feed output rows that the store skips.
    for (int i = tid; i < 512; i += 256) {
      int r  = i >> 2;
      int kc = (i & 3) << 3;
      int gr = bm + r;
      if (gr >= M) gr = M - 1;
      unsigned ldsa = (unsigned)(uintptr_t)&As[r][kc];
      const _Float16* ga = A + (size_t)gr * K + k0 + kc;
      asm volatile("global_load_async_to_lds_b128 %0, %1, off"
                   :: "v"(ldsa), "v"(ga) : "memory");
      if (k0 + 32 < K) __builtin_prefetch(ga + 32, 0, 0);   // next K-tile, same row
    }
    // Stage B tile transposed: B[k0+k][bn+n] -> Bs[n][k].
    // OOB columns clamped to ldb-8: they only feed output cols >= Nout (store-skipped).
    for (int i = tid; i < 512; i += 256) {
      int k  = i >> 4;          // 0..31
      int nc = (i & 15) << 3;   // 0..120, step 8
      int gn = bn + nc;
      if (gn + 8 > ldb) gn = ldb - 8;           // ldb is a multiple of 8
      const _Float16* gb = B + (size_t)(k0 + k) * ldb + gn;
      uint4 v = *(const uint4*)gb;
      if (k0 + 32 < K) __builtin_prefetch(gb + (size_t)32 * ldb, 0, 0);
      union { uint4 q; _Float16 h[8]; } u; u.q = v;
      for (int j = 0; j < 8; ++j) Bs[nc + j][k] = u.h[j];
    }
    asm volatile("s_wait_asynccnt 0x0" ::: "memory");   // A tile landed in LDS
    __syncthreads();

    // A fragment layout (16-bit A 16x32): lane L (m=L%16):
    //   halves 0..7  = K kbase..kbase+7,  halves 8..15 = K kbase+16..kbase+23, kbase=(L/16)*8
    v16h afrag[2];
    for (int fm = 0; fm < 2; ++fm) {
      int row = waveM * 32 + fm * 16 + lm;
      V16 t;
      t.q[0] = *(const uint4*)&As[row][hs * 8];
      t.q[1] = *(const uint4*)&As[row][hs * 8 + 16];
      afrag[fm] = t.v;
    }
    // B fragment layout (32x16): lane L (n=L%16): halves 0..7 = K kbase..kbase+7, halves 8..15 = K kbase+16..kbase+23, kbase=(L/16)*8
    v16h bfrag[4];
    for (int fn = 0; fn < 4; ++fn) {
      int col = waveN * 64 + fn * 16 + lm;
      V16 t;
      t.q[0] = *(const uint4*)&Bs[col][hs * 8];
      t.q[1] = *(const uint4*)&Bs[col][hs * 8 + 16];
      bfrag[fn] = t.v;
    }

    for (int fm = 0; fm < 2; ++fm)
      for (int fn = 0; fn < 4; ++fn)
        acc[fm][fn] = __builtin_amdgcn_wmma_f32_16x16x32_f16(
            false, afrag[fm], false, bfrag[fn], (short)0, acc[fm][fn], false, false);

    __syncthreads();
  }

  // Store: D layout: lane n = L%16; VGPR v -> m = v + 8*(L/16).
  for (int fm = 0; fm < 2; ++fm) {
    for (int fn = 0; fn < 4; ++fn) {
      int col = bn + waveN * 64 + fn * 16 + lm;
      if (col >= Nout) continue;
      for (int v = 0; v < 8; ++v) {
        int row = bm + waveM * 32 + fm * 16 + hs * 8 + v;
        if (row < M) C[(size_t)row * ldc + col] = acc[fm][fn][v];
      }
    }
  }
}

// ---------------- attention kernels ----------------
__device__ inline int f2key(float f) {
  int b = __float_as_int(f);
  return (b >= 0) ? b : (b ^ 0x7FFFFFFF);
}
__device__ inline float key2f(int k) {
  int b = (k >= 0) ? k : (k ^ 0x7FFFFFFF);
  return __int_as_float(b);
}
__device__ inline float lrelu(float v) { return v > 0.0f ? v : NEG_SLOPE * v; }

// es[n,h] = <h[n,h,:], a_src[h,:]>, ed likewise (single pass over h).
__global__ void attn_scores_kernel(const float* __restrict__ h,
                                   const float* __restrict__ asrc,
                                   const float* __restrict__ adst,
                                   float* __restrict__ es, float* __restrict__ ed,
                                   int Nn, int H, int U) {
  int i = blockIdx.x * blockDim.x + threadIdx.x;
  if (i >= Nn * H) return;
  int n = i / H, hh = i % H;
  const float* hp = h + (size_t)n * H * U + (size_t)hh * U;
  const float* as = asrc + (size_t)hh * U;
  const float* ad = adst + (size_t)hh * U;
  float s = 0.f, d = 0.f;
  for (int u = 0; u < U; ++u) { float v = hp[u]; s += v * as[u]; d += v * ad[u]; }
  es[i] = s; ed[i] = d;
}

__global__ void edge_max_kernel(const int* __restrict__ edges,
                                const float* __restrict__ es, const float* __restrict__ ed,
                                int* __restrict__ mkey, int E, int H) {
  int i = blockIdx.x * blockDim.x + threadIdx.x;
  if (i >= E * H) return;
  int e = i / H, hh = i % H;
  int s = edges[2 * e], d = edges[2 * e + 1];
  float v = lrelu(es[s * H + hh] + ed[d * H + hh]);
  atomicMax(&mkey[d * H + hh], f2key(v));
}

__global__ void finalize_max_kernel(const int* __restrict__ mkey, float* __restrict__ mf, int n) {
  int i = blockIdx.x * blockDim.x + threadIdx.x;
  if (i >= n) return;
  int k = mkey[i];
  mf[i] = (k == (int)0x80000000) ? 0.0f : key2f(k);   // isfinite fixup for empty segments
}

__global__ void edge_denom_kernel(const int* __restrict__ edges,
                                  const float* __restrict__ es, const float* __restrict__ ed,
                                  const float* __restrict__ mf, float* __restrict__ denom,
                                  int E, int H) {
  int i = blockIdx.x * blockDim.x + threadIdx.x;
  if (i >= E * H) return;
  int e = i / H, hh = i % H;
  int s = edges[2 * e], d = edges[2 * e + 1];
  float v = lrelu(es[s * H + hh] + ed[d * H + hh]);
  float p = expf(v - mf[d * H + hh]);
  atomicAdd(&denom[d * H + hh], p);
}

// One wave32 per (edge, head): lanes cover the U dimension (coalesced atomics).
__global__ __launch_bounds__(256)
void edge_aggregate_kernel(const int* __restrict__ edges,
                           const float* __restrict__ es, const float* __restrict__ ed,
                           const float* __restrict__ mf, const float* __restrict__ denom,
                           const float* __restrict__ h, float* __restrict__ agg,
                           int E, int H, int U) {
  int gid  = blockIdx.x * 8 + (threadIdx.x >> 5);
  int lane = threadIdx.x & 31;
  if (gid >= E * H) return;
  int e = gid / H, hh = gid % H;
  int s = edges[2 * e], d = edges[2 * e + 1];
  float v = lrelu(es[s * H + hh] + ed[d * H + hh]);
  float alpha = expf(v - mf[d * H + hh]) / fmaxf(denom[d * H + hh], 1e-9f);
  const float* hs = h   + (size_t)s * H * U + (size_t)hh * U;
  float*       od = agg + (size_t)d * H * U + (size_t)hh * U;
  for (int u = lane; u < U; u += 32) atomicAdd(&od[u], alpha * hs[u]);
}

__global__ void add_elu_kernel(const float* __restrict__ a, const float* __restrict__ b,
                               float* __restrict__ o, int n) {
  int i = blockIdx.x * blockDim.x + threadIdx.x;
  if (i >= n) return;
  float z = a[i] + b[i];
  o[i] = z > 0.0f ? z : expm1f(z);
}

__global__ void avg_heads_add_kernel(const float* __restrict__ agg, const float* __restrict__ res,
                                     float* __restrict__ out, int Nn, int H, int U) {
  int i = blockIdx.x * blockDim.x + threadIdx.x;
  if (i >= Nn * U) return;
  int n = i / U, u = i % U;
  float s = 0.f;
  for (int hh = 0; hh < H; ++hh) s += agg[(size_t)n * H * U + (size_t)hh * U + u];
  out[i] = s / (float)H + res[(size_t)n * U + u];
}

// ---------------- host-side helpers ----------------
static void launch_gemm(hipStream_t st, const _Float16* A, const _Float16* B, float* C,
                        int M, int Nout, int K, int ldb, int ldc) {
  dim3 grid(ceil_div_i(M, 128), ceil_div_i(Nout, 128));
  gemm_f16_wmma_kernel<<<grid, 256, 0, st>>>(A, B, C, M, Nout, K, ldb, ldc);
}

static void run_attention(hipStream_t st, const float* h, const float* asrc, const float* adst,
                          const int* edges, int E, int Nn, int H, int U,
                          float* es, float* ed, int* mkey, float* mf, float* denom, float* agg) {
  int nh = Nn * H;
  attn_scores_kernel<<<ceil_div_i(nh, 256), 256, 0, st>>>(h, asrc, adst, es, ed, Nn, H, U);
  fill_i32_kernel<<<ceil_div_i(nh, 256), 256, 0, st>>>(mkey, (int)0x80000000, nh);
  fill_f32_kernel<<<ceil_div_i(nh, 256), 256, 0, st>>>(denom, 0.0f, nh);
  int na = Nn * H * U;
  fill_f32_kernel<<<ceil_div_i(na, 256), 256, 0, st>>>(agg, 0.0f, na);
  int eh = E * H;
  edge_max_kernel<<<ceil_div_i(eh, 256), 256, 0, st>>>(edges, es, ed, mkey, E, H);
  finalize_max_kernel<<<ceil_div_i(nh, 256), 256, 0, st>>>(mkey, mf, nh);
  edge_denom_kernel<<<ceil_div_i(eh, 256), 256, 0, st>>>(edges, es, ed, mf, denom, E, H);
  edge_aggregate_kernel<<<ceil_div_i(eh, 8), 256, 0, st>>>(edges, es, ed, mf, denom, h, agg, E, H, U);
}

extern "C" void kernel_launch(void* const* d_in, const int* in_sizes, int n_in,
                              void* d_out, int out_size, void* d_ws, size_t ws_size,
                              hipStream_t stream) {
  (void)n_in; (void)out_size; (void)ws_size;
  const float* x0    = (const float*)d_in[0];
  const int*   edges = (const int*)d_in[1];
  const float* W1 = (const float*)d_in[2];
  const float* a1s = (const float*)d_in[3];
  const float* a1d = (const float*)d_in[4];
  const float* R1 = (const float*)d_in[5];
  const float* W2 = (const float*)d_in[6];
  const float* a2s = (const float*)d_in[7];
  const float* a2d = (const float*)d_in[8];
  const float* W3 = (const float*)d_in[9];
  const float* a3s = (const float*)d_in[10];
  const float* a3d = (const float*)d_in[11];
  const float* R3 = (const float*)d_in[12];
  float* out = (float*)d_out;

  const int Nn = in_sizes[0] / 128;   // 50000
  const int E  = in_sizes[1] / 2;     // 800000

  // Workspace arena (256B aligned carve-outs).
  char* p = (char*)d_ws;
  auto carve = [&](size_t bytes) {
    void* r = (void*)p;
    p += (bytes + 255) & ~(size_t)255;
    return r;
  };
  _Float16* xf16 = (_Float16*)carve((size_t)Nn * 512 * sizeof(_Float16));
  _Float16* wf16 = (_Float16*)carve((size_t)512 * 728 * sizeof(_Float16));
  float* hbuf   = (float*)carve((size_t)Nn * 726 * sizeof(float));
  float* resbuf = (float*)carve((size_t)Nn * 512 * sizeof(float));
  float* agg    = (float*)carve((size_t)Nn * 726 * sizeof(float));
  float* x1     = (float*)carve((size_t)Nn * 512 * sizeof(float));
  float* x2     = (float*)carve((size_t)Nn * 512 * sizeof(float));
  float* es     = (float*)carve((size_t)Nn * 6 * sizeof(float));
  float* ed     = (float*)carve((size_t)Nn * 6 * sizeof(float));
  int*   mkey   = (int*)  carve((size_t)Nn * 6 * sizeof(int));
  float* mf     = (float*)carve((size_t)Nn * 6 * sizeof(float));
  float* denom  = (float*)carve((size_t)Nn * 6 * sizeof(float));

  // ---- Layer 1: K=128, H=4, U=128, concat, residual x@R1, ELU ----
  cvt_f32_to_f16_kernel<<<ceil_div_i(Nn * 128, 256), 256, 0, stream>>>(x0, xf16, Nn * 128);
  cvt_pad_f16_kernel<<<ceil_div_i(128 * 512, 256), 256, 0, stream>>>(W1, wf16, 128, 512, 512);
  launch_gemm(stream, xf16, wf16, hbuf, Nn, 512, 128, 512, 512);
  cvt_pad_f16_kernel<<<ceil_div_i(128 * 512, 256), 256, 0, stream>>>(R1, wf16, 128, 512, 512);
  launch_gemm(stream, xf16, wf16, resbuf, Nn, 512, 128, 512, 512);
  run_attention(stream, hbuf, a1s, a1d, edges, E, Nn, 4, 128, es, ed, mkey, mf, denom, agg);
  add_elu_kernel<<<ceil_div_i(Nn * 512, 256), 256, 0, stream>>>(agg, resbuf, x1, Nn * 512);

  // ---- Layer 2: K=512, H=4, U=128, concat, residual = x, ELU ----
  cvt_f32_to_f16_kernel<<<ceil_div_i(Nn * 512, 256), 256, 0, stream>>>(x1, xf16, Nn * 512);
  cvt_pad_f16_kernel<<<ceil_div_i(512 * 512, 256), 256, 0, stream>>>(W2, wf16, 512, 512, 512);
  launch_gemm(stream, xf16, wf16, hbuf, Nn, 512, 512, 512, 512);
  run_attention(stream, hbuf, a2s, a2d, edges, E, Nn, 4, 128, es, ed, mkey, mf, denom, agg);
  add_elu_kernel<<<ceil_div_i(Nn * 512, 256), 256, 0, stream>>>(agg, x1, x2, Nn * 512);

  // ---- Layer 3: K=512, H=6, U=121, mean over heads, residual x@R3, identity ----
  cvt_f32_to_f16_kernel<<<ceil_div_i(Nn * 512, 256), 256, 0, stream>>>(x2, xf16, Nn * 512);
  cvt_pad_f16_kernel<<<ceil_div_i(512 * 728, 256), 256, 0, stream>>>(W3, wf16, 512, 726, 728);
  launch_gemm(stream, xf16, wf16, hbuf, Nn, 726, 512, 728, 726);
  cvt_pad_f16_kernel<<<ceil_div_i(512 * 128, 256), 256, 0, stream>>>(R3, wf16, 512, 121, 128);
  launch_gemm(stream, xf16, wf16, resbuf, Nn, 121, 512, 128, 121);
  run_attention(stream, hbuf, a3s, a3d, edges, E, Nn, 6, 121, es, ed, mkey, mf, denom, agg);
  avg_heads_add_kernel<<<ceil_div_i(Nn * 121, 256), 256, 0, stream>>>(agg, resbuf, out, Nn, 6, 121);
}
